// MultiHeadSelfAttention_43559558316489
// MI455X (gfx1250) — matrix operand order repaired and checked by reading, not compiled, not run on hardware
//
#include <hip/hip_runtime.h>
#include <hip/hip_bf16.h>

// ---------------------------------------------------------------------------
// MI455X (gfx1250) multi-head self-attention, bf16 WMMA path.
//   B=8, S=1024, D=1024, H=16, HD=64
// Workspace layout (bf16 elements):
//   xb  [8192,1024]          16 MiB
//   wt  [3][1024(n)][1024(k)] 6 MiB   (transposed weights)
//   qb  [B,H,S,HD]           16 MiB
//   kb  [B,H,S,HD]           16 MiB
//   vt  [B,H,HD,S]           16 MiB   (V transposed for contiguous B-frags)
// total ~70 MiB of d_ws.
// ---------------------------------------------------------------------------

typedef __bf16 bf16;
typedef __attribute__((ext_vector_type(8)))  __bf16 bf16x8;
typedef __attribute__((ext_vector_type(16))) __bf16 v16bf;
typedef __attribute__((ext_vector_type(8)))  float  v8f;
typedef __attribute__((ext_vector_type(4)))  float  f32x4;

constexpr int Bc  = 8;
constexpr int Sc  = 1024;
constexpr int Dc  = 1024;
constexpr int Hc  = 16;
constexpr int HDc = 64;

// A-fragment (16x32 bf16, M x K): lane<16 -> row=lane, K {k0..k0+7, k0+16..k0+23}
//                                 lane>=16 -> row=lane-16, K {k0+8..15, k0+24..31}
static __device__ __forceinline__ v16bf load_a_frag(const bf16* __restrict__ p0,
                                                    int ld, int major0, int k0, int lane) {
  const int r15 = lane & 15, half = lane >> 4;
  const bf16* p = p0 + (size_t)(major0 + r15) * ld + k0 + half * 8;
  bf16x8 lo = *(const bf16x8*)(p);
  bf16x8 hi = *(const bf16x8*)(p + 16);
  v16bf f;
#pragma unroll
  for (int i = 0; i < 8; ++i) { f[i] = lo[i]; f[i + 8] = hi[i]; }
  return f;
}

// B-fragment (32x16 bf16, K x N), source stored column-major (col contiguous in k):
//   lane<16 -> col=lane, K {k0..k0+7, k0+16..k0+23};  lane>=16 -> col=lane-16, K {k0+8..15, k0+24..31}
static __device__ __forceinline__ v16bf load_b_frag(const bf16* __restrict__ p0,
                                                    int ld, int major0, int k0, int lane) {
  const int r15 = lane & 15, half = lane >> 4;
  const bf16* p = p0 + (size_t)(major0 + r15) * ld + k0 + half * 8;
  bf16x8 lo = *(const bf16x8*)(p);
  bf16x8 hi = *(const bf16x8*)(p + 16);
  v16bf f;
#pragma unroll
  for (int i = 0; i < 8; ++i) { f[i] = lo[i]; f[i + 8] = hi[i]; }
  return f;
}

static __device__ __forceinline__ v8f wmma_bf16(v16bf a, v16bf b, v8f c) {
  return __builtin_amdgcn_wmma_f32_16x16x32_bf16(false, a, false, b, (short)0, c,
                                                 false, false);
}

// ---------------------------------------------------------------------------
__global__ __launch_bounds__(256) void cvt_x_kernel(const float* __restrict__ x,
                                                    bf16* __restrict__ xb) {
  const size_t i = (size_t)blockIdx.x * blockDim.x + threadIdx.x;
  xb[i] = (bf16)x[i];
}

// wt[z][n][k] = W_z[k][n]  (transpose + downconvert)
__global__ __launch_bounds__(256) void cvt_w_kernel(const float* __restrict__ wq,
                                                    const float* __restrict__ wk,
                                                    const float* __restrict__ wv,
                                                    bf16* __restrict__ wt) {
  const int idx = blockIdx.x * blockDim.x + threadIdx.x;   // over D*D
  const int z = blockIdx.z;
  const int k = idx >> 10;
  const int n = idx & (Dc - 1);
  const float* src = (z == 0) ? wq : (z == 1) ? wk : wv;
  wt[(size_t)z * Dc * Dc + (size_t)n * Dc + k] = (bf16)src[idx];
}

// ---------------------------------------------------------------------------
// QKV projection GEMM: [8192 x 1024] @ [1024 x 1024] per matrix.
// Block = 4 waves; each wave computes one M16 x N64 strip (A reused across 4 B tiles).
// z=0 -> Q [B,H,S,HD], z=1 -> K [B,H,S,HD], z=2 -> V transposed [B,H,HD,S].
__global__ __launch_bounds__(128) void gemm_qkv_kernel(const bf16* __restrict__ xb,
                                                       const bf16* __restrict__ wt_all,
                                                       bf16* __restrict__ qb,
                                                       bf16* __restrict__ kb,
                                                       bf16* __restrict__ vt) {
  const int lane = threadIdx.x & 31;
  const int wave = threadIdx.x >> 5;
  const int z = blockIdx.z;
  const bf16* wt = wt_all + (size_t)z * Dc * Dc;       // [n][k]
  const int m0 = blockIdx.y * 64 + wave * 16;          // row tile in [0,8192)
  const int n0 = blockIdx.x * 64;                      // col group (4 x 16)

  v8f acc[4] = {v8f{}, v8f{}, v8f{}, v8f{}};
  for (int k0 = 0; k0 < Dc; k0 += 32) {
    const v16bf a = load_a_frag(xb, Dc, m0, k0, lane);
#pragma unroll
    for (int j = 0; j < 4; ++j) {
      const v16bf b = load_b_frag(wt, Dc, n0 + j * 16, k0, lane);
      acc[j] = wmma_bf16(a, b, acc[j]);
    }
  }

  const int r15 = lane & 15, half = lane >> 4;
#pragma unroll
  for (int j = 0; j < 4; ++j) {
#pragma unroll
    for (int r = 0; r < 8; ++r) {
      const int row = m0 + half * 8 + r;      // global row in [0,8192)
      const int col = n0 + j * 16 + r15;      // global col in [0,1024)
      const int b_  = row >> 10;              // batch
      const int sl  = row & (Sc - 1);         // seq
      const int h   = col >> 6;               // head
      const int hd  = col & (HDc - 1);        // head dim
      const bf16 v = (bf16)acc[j][r];
      if (z == 0)      qb[(((size_t)(b_ * Hc + h) * Sc + sl) * HDc) + hd] = v;
      else if (z == 1) kb[(((size_t)(b_ * Hc + h) * Sc + sl) * HDc) + hd] = v;
      else             vt[(((size_t)(b_ * Hc + h) * HDc + hd) * Sc) + sl] = v;
    }
  }
}

// ---------------------------------------------------------------------------
// Streaming attention. Unsafe softmax (no max-subtraction) => no rescaling:
//   acc += e_tile @ V_tile ; dsum += rowsum(e_tile) ; out = acc / (dsum + 1e-8)
// One wave handles 16 query rows; block = 4 waves = 64 rows.
__global__ __launch_bounds__(128) void attn_kernel(const bf16* __restrict__ qb,
                                                   const bf16* __restrict__ kb,
                                                   const bf16* __restrict__ vt,
                                                   const float* __restrict__ mask,
                                                   float* __restrict__ out) {
  __shared__ __align__(16) bf16  e_lds[4][16][40];   // bf16 e tile, padded rows (80B)
  __shared__ __align__(16) float red[4][16][16];     // denominator reduction tile

  const int lane = threadIdx.x & 31;
  const int wave = threadIdx.x >> 5;
  const int bh = blockIdx.x;                 // 0..127
  const int b  = bh >> 4;
  const int h  = bh & (Hc - 1);
  const int s0 = blockIdx.y * 64 + wave * 16;

  const bf16* Qp = qb + (size_t)bh * Sc * HDc;   // [S][HD]
  const bf16* Kp = kb + (size_t)bh * Sc * HDc;   // [S][HD]
  const bf16* Vp = vt + (size_t)bh * HDc * Sc;   // [HD][S]
  const float* Mp = mask + (size_t)b * Sc * Sc;  // [S][S]

  const int r15 = lane & 15, half = lane >> 4;

  // Q A-fragments for K-dims 0..31 and 32..63 (reused for all key tiles)
  const v16bf aq0 = load_a_frag(Qp, HDc, s0, 0, lane);
  const v16bf aq1 = load_a_frag(Qp, HDc, s0, 32, lane);

  v8f acc[4] = {v8f{}, v8f{}, v8f{}, v8f{}};
  float dsum[8] = {0.f, 0.f, 0.f, 0.f, 0.f, 0.f, 0.f, 0.f};

  for (int tb = 0; tb < Sc; tb += 32) {
    if (tb + 32 < Sc) {
      __builtin_prefetch(Kp + (size_t)(tb + 32) * HDc, 0, 0);  // global_prefetch_b8
      __builtin_prefetch(Vp + (size_t)(tb + 32), 0, 0);
    }
    // --- scores + exp*mask for two 16-key subtiles, staged to LDS as bf16 ---
#pragma unroll
    for (int tt = 0; tt < 2; ++tt) {
      const int t0 = tb + tt * 16;
      const v16bf bk0 = load_b_frag(Kp, HDc, t0, 0, lane);
      const v16bf bk1 = load_b_frag(Kp, HDc, t0, 32, lane);
      v8f c = {};
      c = wmma_bf16(aq0, bk0, c);
      c = wmma_bf16(aq1, bk1, c);
#pragma unroll
      for (int r = 0; r < 8; ++r) {
        const int m   = half * 8 + r;
        const int col = t0 + r15;
        const float e = __expf(c[r] * 0.125f) *
                        Mp[(size_t)(s0 + m) * Sc + col];
        dsum[r] += e;                                  // f32 partial row-sums
        e_lds[wave][m][tt * 16 + r15] = (bf16)e;       // stage for relayout
      }
    }
    __builtin_amdgcn_wave_barrier();   // order ds stores before cross-lane ds loads

    // --- reload e as an A-fragment (16 x 32, D-layout -> A-layout transpose) ---
    v16bf ae;
    {
      const bf16* p = &e_lds[wave][r15][half * 8];
      bf16x8 lo = *(const bf16x8*)(p);
      bf16x8 hi = *(const bf16x8*)(p + 16);
#pragma unroll
      for (int i = 0; i < 8; ++i) { ae[i] = lo[i]; ae[i + 8] = hi[i]; }
    }
    __builtin_amdgcn_wave_barrier();   // keep next iteration's stores after loads

    // --- PV: e(16x32) @ V(32x64) accumulated over all key tiles ---
#pragma unroll
    for (int j = 0; j < 4; ++j) {
      const v16bf bv = load_b_frag(Vp, Sc, j * 16, tb, lane);
      acc[j] = wmma_bf16(ae, bv, acc[j]);
    }
  }

  // --- denominator: cross-lane reduce the D-layout partial sums via LDS ---
#pragma unroll
  for (int r = 0; r < 8; ++r) red[wave][half * 8 + r][r15] = dsum[r];
  __builtin_amdgcn_wave_barrier();

  float denom[8];
#pragma unroll
  for (int r = 0; r < 8; ++r) {
    const int m = half * 8 + r;
    const f32x4 a0 = *(const f32x4*)&red[wave][m][0];
    const f32x4 a1 = *(const f32x4*)&red[wave][m][4];
    const f32x4 a2 = *(const f32x4*)&red[wave][m][8];
    const f32x4 a3 = *(const f32x4*)&red[wave][m][12];
    float s = 1e-8f;
#pragma unroll
    for (int i = 0; i < 4; ++i) s += a0[i] + a1[i] + a2[i] + a3[i];
    denom[r] = s;
  }

  // --- divide and write out[B,S,D] (fp32) ---
#pragma unroll
  for (int j = 0; j < 4; ++j) {
#pragma unroll
    for (int r = 0; r < 8; ++r) {
      const int m  = half * 8 + r;
      const int hd = j * 16 + r15;
      out[(size_t)(b * Sc + s0 + m) * Dc + h * HDc + hd] = acc[j][r] / denom[r];
    }
  }
}

// ---------------------------------------------------------------------------
extern "C" void kernel_launch(void* const* d_in, const int* in_sizes, int n_in,
                              void* d_out, int out_size, void* d_ws, size_t ws_size,
                              hipStream_t stream) {
  (void)in_sizes; (void)n_in; (void)out_size; (void)ws_size;
  const float* x    = (const float*)d_in[0];
  const float* mask = (const float*)d_in[1];
  const float* Wq   = (const float*)d_in[2];
  const float* Wk   = (const float*)d_in[3];
  const float* Wv   = (const float*)d_in[4];
  float* out = (float*)d_out;

  bf16* xb = (bf16*)d_ws;                               // 8192*1024
  bf16* wt = xb + (size_t)8192 * 1024;                  // 3*1024*1024
  bf16* qb = wt + (size_t)3 * 1024 * 1024;              // 8192*1024
  bf16* kb = qb + (size_t)8192 * 1024;                  // 8192*1024
  bf16* vt = kb + (size_t)8192 * 1024;                  // 8192*1024

  cvt_x_kernel<<<dim3(32768), dim3(256), 0, stream>>>(x, xb);
  cvt_w_kernel<<<dim3(4096, 1, 3), dim3(256), 0, stream>>>(Wq, Wk, Wv, wt);
  gemm_qkv_kernel<<<dim3(16, 128, 3), dim3(128), 0, stream>>>(xb, wt, qb, kb, vt);
  attn_kernel<<<dim3(128, 16), dim3(128), 0, stream>>>(qb, kb, vt, mask, out);
}
